// RelativeMultiHeadAttention_52750788329614
// MI455X (gfx1250) — matrix operand order repaired and checked by reading, not compiled, not run on hardware
//
#include <hip/hip_runtime.h>

#define D_MODEL 512
#define SEQ     64
#define NHEADS  16
#define DK      32
#define LDX     520   // row stride (bf16 elems) for 64x512 tiles, +8 pad
#define LDVT    72    // row stride for 512x64 V^T, +8 pad
#define LDP     72    // row stride for 16x64 per-wave P scratch

typedef __attribute__((ext_vector_type(16))) __bf16 v16bf;
typedef __attribute__((ext_vector_type(8)))  float  v8f;
typedef __attribute__((ext_vector_type(4)))  __bf16 v4bf;

union Frag { v16bf v; uint4 q[2]; };

__device__ __forceinline__ v8f wmma_bf16(const Frag& a, const Frag& b, v8f c) {
  return __builtin_amdgcn_wmma_f32_16x16x32_bf16(false, a.v, false, b.v,
                                                 (short)0, c, false, false);
}

// A fragment (16x32 bf16): lane holds row (lane&15); two 8-elem chunks at
// k + laneHi*8 and k + laneHi*8 + 16 (per ISA 16-bit A layout).
__device__ __forceinline__ void loadA(Frag& a, const __bf16* base, int row,
                                      int ld, int k, int laneHi) {
  const __bf16* p = base + row * ld + k + laneHi * 8;
  a.q[0] = *(const uint4*)(p);
  a.q[1] = *(const uint4*)(p + 16);
}

// B fragment (32x16 bf16): lane holds column (lane&15); two 8-elem chunks at
// k + laneHi*8 and k + laneHi*8 + 16. "col" indexes a row of a row-major [N][K] matrix
// (i.e. B = that matrix transposed), which matches y = x @ W^T directly.
__device__ __forceinline__ void loadB(Frag& b, const __bf16* base, int col,
                                      int ld, int k, int laneHi) {
  const __bf16* p = base + col * ld + k + laneHi * 8;
  b.q[0] = *(const uint4*)(p);
  b.q[1] = *(const uint4*)(p + 16);
}

// ---- weight pre-convert: 4x (512x512) f32 -> bf16 into workspace ----------
__global__ void convert_w(const float* __restrict__ w0, const float* __restrict__ w1,
                          const float* __restrict__ w2, const float* __restrict__ w3,
                          __bf16* __restrict__ dst) {
  int i4   = blockIdx.x * blockDim.x + threadIdx.x;   // 0 .. 262143
  int base = i4 * 4;                                  // element index
  int p    = base >> 18;                              // which matrix (2^18 elems each)
  int off  = base & 262143;
  const float* src = (p == 0) ? w0 : (p == 1) ? w1 : (p == 2) ? w2 : w3;
  float4 v = ((const float4*)src)[off >> 2];
  v4bf o;
  o.x = (__bf16)v.x; o.y = (__bf16)v.y; o.z = (__bf16)v.z; o.w = (__bf16)v.w;
  *(v4bf*)&dst[base] = o;
}

// ---- fused MHA: one workgroup per batch row --------------------------------
__global__ __launch_bounds__(512, 1)
void fused_mha(const float* __restrict__ x, const __bf16* __restrict__ Wb,
               float* __restrict__ out) {
  __shared__ __bf16 sX [SEQ * LDX];       // x_b (bf16), reused as attention out O
  __shared__ __bf16 sQ [SEQ * LDX];
  __shared__ __bf16 sK [SEQ * LDX];
  __shared__ __bf16 sVT[D_MODEL * LDVT];  // V transposed: [channel][key]
  __shared__ __bf16 sP [NHEADS * 16 * LDP]; // per-wave softmax(P) scratch

  const int tid    = threadIdx.x;
  const int wave   = tid >> 5;
  const int lane   = tid & 31;
  const int laneN  = lane & 15;
  const int laneHi = lane >> 4;

  const float* xb = x + (size_t)blockIdx.x * (SEQ * D_MODEL);

  // Phase 1: stage x_b into LDS as bf16
  for (int i = 0; i < 16; ++i) {
    int f4 = tid + i * 512;                 // float4 index, 8192 total
    int r  = f4 >> 7;                       // 128 float4 per row
    int c  = (f4 & 127) << 2;
    float4 v = ((const float4*)xb)[f4];
    v4bf o;
    o.x = (__bf16)v.x; o.y = (__bf16)v.y; o.z = (__bf16)v.z; o.w = (__bf16)v.w;
    *(v4bf*)&sX[r * LDX + c] = o;
  }
  __syncthreads();

  const int mt     = wave & 3;              // 4 row tiles of 16
  const int ntBase = (wave >> 2) * 8;       // 8 col tiles per wave
  const int aRow   = mt * 16 + laneN;

  // Phase 2: Q, K, V projections (M=64, N=512, K=512)
  for (int p = 0; p < 3; ++p) {
    const __bf16* W = Wb + p * (D_MODEL * D_MODEL);
    for (int j = 0; j < 8; ++j) {
      int nt = ntBase + j;
      v8f c = {0.f,0.f,0.f,0.f,0.f,0.f,0.f,0.f};
      #pragma unroll 4
      for (int kt = 0; kt < 16; ++kt) {
        Frag a, b;
        loadA(a, sX, aRow, LDX, kt * 32, laneHi);
        loadB(b, W, nt * 16 + laneN, D_MODEL, kt * 32, laneHi);
        c = wmma_bf16(a, b, c);
      }
      if (p == 2) {                          // V: store transposed
        int dchan = nt * 16 + laneN;
        #pragma unroll
        for (int r2 = 0; r2 < 8; ++r2) {
          int m = mt * 16 + r2 + laneHi * 8;
          sVT[dchan * LDVT + m] = (__bf16)c[r2];
        }
      } else {
        __bf16* dst = (p == 0) ? sQ : sK;
        int col = nt * 16 + laneN;
        #pragma unroll
        for (int r2 = 0; r2 < 8; ++r2) {
          int m = mt * 16 + r2 + laneHi * 8;
          dst[m * LDX + col] = (__bf16)c[r2];
        }
      }
    }
  }
  __syncthreads();

  // Phase 3: attention; wave w owns head h = w
  const int h = wave;
  __bf16* sPw = &sP[wave * (16 * LDP)];
  const float scale = 0.17677669529663687f;  // 1/sqrt(32)
  for (int mt2 = 0; mt2 < 4; ++mt2) {
    v8f sc[4];
    #pragma unroll
    for (int t = 0; t < 4; ++t) {
      Frag a, b;
      loadA(a, sQ, mt2 * 16 + laneN, LDX, h * DK, laneHi);
      loadB(b, sK, t * 16 + laneN, LDX, h * DK, laneHi);
      v8f z = {0.f,0.f,0.f,0.f,0.f,0.f,0.f,0.f};
      sc[t] = wmma_bf16(a, b, z);            // Q(16x32) @ K^T(32x16)
    }
    // softmax per row (rows live in 16-lane halves; VGPR r -> row r (+8 upper))
    #pragma unroll
    for (int r2 = 0; r2 < 8; ++r2) {
      float mx = fmaxf(fmaxf(sc[0][r2], sc[1][r2]), fmaxf(sc[2][r2], sc[3][r2]));
      for (int m2 = 1; m2 < 16; m2 <<= 1) mx = fmaxf(mx, __shfl_xor(mx, m2, 32));
      float s = 0.f;
      #pragma unroll
      for (int t = 0; t < 4; ++t) {
        float e = __expf((sc[t][r2] - mx) * scale);
        sc[t][r2] = e; s += e;
      }
      for (int m2 = 1; m2 < 16; m2 <<= 1) s += __shfl_xor(s, m2, 32);
      float inv = 1.0f / s;
      #pragma unroll
      for (int t = 0; t < 4; ++t) sc[t][r2] *= inv;
    }
    // re-layout P via per-wave LDS scratch (row-major 16x64 bf16)
    #pragma unroll
    for (int t = 0; t < 4; ++t)
      #pragma unroll
      for (int r2 = 0; r2 < 8; ++r2) {
        int m = r2 + laneHi * 8;
        sPw[m * LDP + t * 16 + laneN] = (__bf16)sc[t][r2];
      }
    // O(16x32) = P(16x64) @ V(64x32)
    #pragma unroll
    for (int ot = 0; ot < 2; ++ot) {
      v8f c = {0.f,0.f,0.f,0.f,0.f,0.f,0.f,0.f};
      #pragma unroll
      for (int kt = 0; kt < 2; ++kt) {
        Frag a, b;
        loadA(a, sPw, laneN, LDP, kt * 32, laneHi);
        loadB(b, sVT, h * DK + ot * 16 + laneN, LDVT, kt * 32, laneHi);
        c = wmma_bf16(a, b, c);
      }
      int col = h * DK + ot * 16 + laneN;
      #pragma unroll
      for (int r2 = 0; r2 < 8; ++r2) {
        int m = mt2 * 16 + r2 + laneHi * 8;
        sX[m * LDX + col] = (__bf16)c[r2];   // sX now holds O
      }
    }
  }
  __syncthreads();

  // Phase 4: Y = O @ Wo^T -> global f32
  const __bf16* Wo = Wb + 3 * (D_MODEL * D_MODEL);
  float* ob = out + (size_t)blockIdx.x * (SEQ * D_MODEL);
  for (int j = 0; j < 8; ++j) {
    int nt = ntBase + j;
    v8f c = {0.f,0.f,0.f,0.f,0.f,0.f,0.f,0.f};
    #pragma unroll 4
    for (int kt = 0; kt < 16; ++kt) {
      Frag a, b;
      loadA(a, sX, aRow, LDX, kt * 32, laneHi);
      loadB(b, Wo, nt * 16 + laneN, D_MODEL, kt * 32, laneHi);
      c = wmma_bf16(a, b, c);
    }
    int col = nt * 16 + laneN;
    #pragma unroll
    for (int r2 = 0; r2 < 8; ++r2) {
      int m = mt * 16 + r2 + laneHi * 8;
      ob[m * D_MODEL + col] = c[r2];
    }
  }
}

extern "C" void kernel_launch(void* const* d_in, const int* in_sizes, int n_in,
                              void* d_out, int out_size, void* d_ws, size_t ws_size,
                              hipStream_t stream) {
  const float* x  = (const float*)d_in[0];
  const float* Wq = (const float*)d_in[1];
  const float* Wk = (const float*)d_in[2];
  const float* Wv = (const float*)d_in[3];
  const float* Wo = (const float*)d_in[4];
  __bf16* Wb = (__bf16*)d_ws;   // needs 4 * 512 * 512 * 2 = 2 MB of workspace

  convert_w<<<1024, 256, 0, stream>>>(Wq, Wk, Wv, Wo, Wb);
  fused_mha<<<2048, 512, 0, stream>>>(x, Wb, (float*)d_out);
}
